// AdvKPCN_5050881540209
// MI455X (gfx1250) — compile-verified, matrix operand order checked by reading, not run on hardware
//
#include <hip/hip_runtime.h>
#include <hip/hip_bf16.h>
#include <cstdint>
#include <cstddef>

typedef __attribute__((ext_vector_type(16))) _Float16 v16h;
typedef __attribute__((ext_vector_type(2)))  _Float16 h2v;
typedef __attribute__((ext_vector_type(8)))  float    v8f;

#define CONV_BLOCK 256
#define BLOCK_PIX 128          /* 8 waves x 16 pixels */
#define TILE_MBLK 32           /* couts per block (2 x 16 m-subtiles) */

// ---------------------------------------------------------------------------
// Implicit-GEMM conv, f16 WMMA, f32 accumulate.
// Block: 256 threads = 8 wave32; block tile = 32 cout x 128 pixels.
// K-step = 64: per wave, ALL 6 fragments (2 B + 4 A) are preloaded as 32B
// vectors (12 x ds_load_b128), then 4 x V_WMMA_F32_16X16X32_F16 issue
// back-to-back (single ds wait instead of one per WMMA).
// LDS tiles stored PRE-PERMUTED into the ISA 16-bit fragment layout:
//   element kk (0..31) of row r lives at [lane = r + 16*h][i], where
//   h = (kk>>3)&1, kk2 = kk-8h, i = kk2<8 ? kk2 : kk2-8.
// Double-buffered LDS; global loads for step s+1 overlap WMMAs of step s.
// Weights are pre-packed to f16 (pack_f16) once per launch.
// ---------------------------------------------------------------------------
template <int KH, int PAD>
__global__ __launch_bounds__(CONV_BLOCK)
void conv2d_wmma(const float* __restrict__ x, const _Float16* __restrict__ w,
                 const float* __restrict__ bias, float* __restrict__ y,
                 int Cin, int Hin, int Win,
                 int Cout, int Hout, int Wout, int relu)
{
    constexpr int KK = KH * KH;
    __shared__ alignas(32) _Float16 sA[2][2][2][32][16];    // [buf][ksub][msub][lane][elem]  8 KB
    __shared__ alignas(32) _Float16 sB[2][2][8][32][16];    // [buf][ksub][wave][lane][elem] 32 KB

    const int t    = threadIdx.x;
    const int lane = t & 31;
    const int wid  = t >> 5;
    const int b    = blockIdx.z;
    const int m0   = blockIdx.y * TILE_MBLK;
    const int pix0 = blockIdx.x * BLOCK_PIX;
    const int Npix = Hout * Wout;
    const int Ktot = Cin * KK;

    const float* xb = x + (size_t)b * Cin * Hin * Win;

    // K-invariant staging geometry
    const int  nB  = t & 127;          // pixel column owned by this thread
    const int  hiB = t >> 7;
    const int  pB  = pix0 + nB;
    const bool pOK = pB < Npix;
    const int  oyB = pB / Wout;
    const int  oxB = pB - oyB * Wout;
    const int  mA  = t >> 4;           // cout row (within 16) owned for A staging
    const int  aA  = t & 15;           // K-pair index

    v8f acc[2] = {};

    auto loadA = [&](int k0, _Float16 (&va)[8]) {
        #pragma unroll
        for (int ms = 0; ms < 2; ++ms) {
            const bool   mOK = (m0 + ms * 16 + mA) < Cout;
            const size_t row = (size_t)(m0 + ms * 16 + mA) * Ktot;
            #pragma unroll
            for (int ks = 0; ks < 2; ++ks)
                #pragma unroll
                for (int q = 0; q < 2; ++q) {
                    int gk = k0 + 32 * ks + 2 * aA + q;
                    _Float16 v = (_Float16)0.f;
                    if (mOK && gk < Ktot) v = w[row + gk];
                    va[ms * 4 + ks * 2 + q] = v;
                }
        }
    };
    auto loadB = [&](int k0, float (&vb)[32]) {
        #pragma unroll
        for (int ks = 0; ks < 2; ++ks)
            #pragma unroll
            for (int j = 0; j < 8; ++j)
                #pragma unroll
                for (int q = 0; q < 2; ++q) {
                    int kk = 2 * (hiB + 2 * j) + q;
                    int gk = k0 + 32 * ks + kk;
                    float v = 0.f;
                    if (pOK && gk < Ktot) {
                        int ci = gk / KK;              // const divide -> mul/shift
                        int r  = gk - ci * KK;
                        int ky = r / KH, kx = r - ky * KH;
                        int iy = oyB + ky - PAD, ix = oxB + kx - PAD;
                        if (PAD == 0 || (iy >= 0 && iy < Hin && ix >= 0 && ix < Win))
                            v = xb[((size_t)ci * Hin + iy) * Win + ix];
                    }
                    vb[ks * 16 + 2 * j + q] = v;
                }
    };
    auto stage = [&](int buf, const _Float16 (&va)[8], const float (&vb)[32]) {
        {   // A pairs at kk = 2*aA
            int kk = 2 * aA;
            int h = (kk >> 3) & 1, kk2 = kk - 8 * h;
            int i = (kk2 < 8) ? kk2 : kk2 - 8;
            #pragma unroll
            for (int ms = 0; ms < 2; ++ms)
                #pragma unroll
                for (int ks = 0; ks < 2; ++ks) {
                    h2v pk; pk.x = va[ms * 4 + ks * 2]; pk.y = va[ms * 4 + ks * 2 + 1];
                    *(h2v*)&sA[buf][ks][ms][mA + 16 * h][i] = pk;
                }
        }
        #pragma unroll
        for (int ks = 0; ks < 2; ++ks)
            #pragma unroll
            for (int j = 0; j < 8; ++j) {   // B pairs
                int kk = 2 * (hiB + 2 * j);
                int h = (kk >> 3) & 1, kk2 = kk - 8 * h;
                int i = (kk2 < 8) ? kk2 : kk2 - 8;
                h2v pk; pk.x = (_Float16)vb[ks * 16 + 2 * j];
                        pk.y = (_Float16)vb[ks * 16 + 2 * j + 1];
                *(h2v*)&sB[buf][ks][nB >> 4][(nB & 15) + 16 * h][i] = pk;
            }
    };
    auto compute = [&](int buf) {
        // Preload ALL fragments, then issue the 4 WMMAs back-to-back.
        v16h bf[2], af[2][2];
        #pragma unroll
        for (int ks = 0; ks < 2; ++ks) {
            bf[ks]    = *(const v16h*)&sB[buf][ks][wid][lane][0];
            af[ks][0] = *(const v16h*)&sA[buf][ks][0][lane][0];
            af[ks][1] = *(const v16h*)&sA[buf][ks][1][lane][0];
        }
        #pragma unroll
        for (int ks = 0; ks < 2; ++ks)
            #pragma unroll
            for (int ms = 0; ms < 2; ++ms)
                acc[ms] = __builtin_amdgcn_wmma_f32_16x16x32_f16(
                              false, af[ks][ms], false, bf[ks], (short)0, acc[ms], false, false);
    };

    const int nsteps = (Ktot + 63) / 64;
    _Float16 va[8]; float vb[32];
    loadA(0, va); loadB(0, vb); stage(0, va, vb);
    __syncthreads();

    for (int s = 0; s < nsteps; ++s) {
        int cur = s & 1;
        if (s + 1 < nsteps) {
            __builtin_prefetch(&w[(size_t)(m0 + mA) * Ktot + 64 * (s + 1)], 0, 3);
            loadA(64 * (s + 1), va);      // global loads in flight ...
            loadB(64 * (s + 1), vb);
        }
        compute(cur);                     // ... while WMMAs run
        if (s + 1 < nsteps) stage(cur ^ 1, va, vb);
        __syncthreads();
    }

    // Epilogue: C layout (VGPR v: lane<16 -> M=v, lane>=16 -> M=v+8; N=lane&15)
    const int h = lane >> 4;
    const int n = pix0 + wid * 16 + (lane & 15);
    #pragma unroll
    for (int ms = 0; ms < 2; ++ms)
        #pragma unroll
        for (int v = 0; v < 8; ++v) {
            int m = m0 + ms * 16 + v + 8 * h;
            if (m < Cout && n < Npix) {
                float r = acc[ms][v] + bias[m];
                if (relu) r = fmaxf(r, 0.f);
                y[((size_t)b * Cout + m) * Npix + n] = r;
            }
        }
}

__global__ __launch_bounds__(256)
void pack_f16(const float* __restrict__ in, _Float16* __restrict__ out, int n)
{
    int i = blockIdx.x * blockDim.x + threadIdx.x;
    if (i < n) out[i] = (_Float16)in[i];
}

// ---------------------------------------------------------------------------
// Softmax(441)-weighted 21x21 gather
// ---------------------------------------------------------------------------
__global__ __launch_bounds__(256)
void kernel_apply_k(const float* __restrict__ data, const float* __restrict__ kern,
                    float* __restrict__ out, int B, int H, int W)
{
    int idx = blockIdx.x * blockDim.x + threadIdx.x;
    int Np  = H * W;
    if (idx >= B * Np) return;
    int b = idx / Np, p = idx - b * Np;
    int yy = p / W, xx = p - yy * W;
    const float* kb = kern + (size_t)b * 441 * Np + p;
    float mx = -1e30f;
    for (int k = 0; k < 441; ++k) mx = fmaxf(mx, kb[(size_t)k * Np]);
    const float* db = data + (size_t)b * 3 * Np;
    float se = 0.f, n0 = 0.f, n1 = 0.f, n2 = 0.f;
    for (int k = 0; k < 441; ++k) {
        float e = __expf(kb[(size_t)k * Np] - mx);
        se += e;
        int ky = k / 21, kx = k - ky * 21;
        int iy = yy + ky - 10, ix = xx + kx - 10;
        if (iy >= 0 && iy < H && ix >= 0 && ix < W) {
            int q = iy * W + ix;
            n0 += e * db[q];
            n1 += e * db[Np + q];
            n2 += e * db[2 * Np + q];
        }
    }
    float inv = 1.f / se;
    float* ob = out + (size_t)b * 3 * Np + p;
    ob[0] = n0 * inv; ob[Np] = n1 * inv; ob[2 * Np] = n2 * inv;
}

__global__ __launch_bounds__(256)
void softmax_ch(const float* __restrict__ in, float* __restrict__ out,
                int B, int C, int Np)
{
    int i = blockIdx.x * blockDim.x + threadIdx.x;
    if (i >= B * Np) return;
    int b = i / Np, p = i - b * Np;
    const float* ib = in + (size_t)b * C * Np + p;
    float mx = -1e30f;
    for (int c = 0; c < C; ++c) mx = fmaxf(mx, ib[(size_t)c * Np]);
    float se = 0.f;
    for (int c = 0; c < C; ++c) se += __expf(ib[(size_t)c * Np] - mx);
    float inv = 1.f / se;
    float* ob = out + (size_t)b * C * Np + p;
    for (int c = 0; c < C; ++c)
        ob[(size_t)c * Np] = __expf(ib[(size_t)c * Np] - mx) * inv;
}

__global__ __launch_bounds__(256)
void maxpool2(const float* __restrict__ in, float* __restrict__ out,
              int B, int C, int H, int W)
{
    int Hh = H >> 1, Wh = W >> 1;
    int i = blockIdx.x * blockDim.x + threadIdx.x;
    if (i >= B * C * Hh * Wh) return;
    int x = i % Wh; int t = i / Wh;
    int y = t % Hh; t /= Hh;
    int c = t % C;  int b = t / C;
    const float* p = in + (((size_t)b * C + c) * H + 2 * y) * W + 2 * x;
    out[i] = fmaxf(fmaxf(p[0], p[1]), fmaxf(p[W], p[W + 1]));
}

__global__ __launch_bounds__(256)
void upsample2(const float* __restrict__ in, float* __restrict__ out,
               int B, int C, int H, int W)   // H,W = OUTPUT dims
{
    int i = blockIdx.x * blockDim.x + threadIdx.x;
    if (i >= B * C * H * W) return;
    int x = i % W; int t = i / W;
    int y = t % H; t /= H;
    int c = t % C; int b = t / C;
    out[i] = in[(((size_t)b * C + c) * (H >> 1) + (y >> 1)) * (W >> 1) + (x >> 1)];
}

__global__ __launch_bounds__(256)
void sigmoid_ch0(const float* __restrict__ in, float* __restrict__ out,
                 int B, int C, int Np)
{
    int i = blockIdx.x * blockDim.x + threadIdx.x;
    if (i >= B * Np) return;
    int b = i / Np, p = i - b * Np;
    float v = in[(size_t)b * C * Np + p];
    out[i] = 1.f / (1.f + __expf(-v));
}

// dst[b, c0d+c, y, x] = src[b, c0s+c, y+dh, x+dw]
__global__ __launch_bounds__(256)
void copy_crop(const float* __restrict__ src, float* __restrict__ dst,
               int B, int Cs, int Hs, int Ws, int c0s, int nc,
               int dh, int dw, int Cd, int Hd, int Wd, int c0d)
{
    int i = blockIdx.x * blockDim.x + threadIdx.x;
    if (i >= B * nc * Hd * Wd) return;
    int x = i % Wd; int t = i / Wd;
    int y = t % Hd; t /= Hd;
    int c = t % nc; int b = t / nc;
    dst[(((size_t)b * Cd + c0d + c) * Hd + y) * Wd + x] =
        src[(((size_t)b * Cs + c0s + c) * Hs + y + dh) * Ws + x + dw];
}

__global__ __launch_bounds__(256)
void combine_radiance(const float* __restrict__ albedo, int Ha, int Wa, int off,
                      const float* __restrict__ rd, const float* __restrict__ rs,
                      float* __restrict__ out, int B, int H, int W)
{
    int i = blockIdx.x * blockDim.x + threadIdx.x;
    if (i >= B * 3 * H * W) return;
    int x = i % W; int t = i / W;
    int y = t % H; t /= H;
    int c = t % 3; int b = t / 3;
    float a = albedo[(((size_t)b * 3 + c) * Ha + y + off) * Wa + x + off];
    size_t q = (((size_t)b * 3 + c) * H + y) * W + x;
    out[q] = a * rd[q] + __expf(rs[q]) - 1.f;
}

// k[b,c,p] += kg[b,c,p] * fake[b,0,p]
__global__ __launch_bounds__(256)
void kernel_fma_fake(float* __restrict__ k, const float* __restrict__ kg,
                     const float* __restrict__ fake, int B, int C, int Np)
{
    int i = blockIdx.x * blockDim.x + threadIdx.x;
    if (i >= B * C * Np) return;
    int p = i % Np;
    int b = (i / Np) / C;
    k[i] += kg[i] * fake[(size_t)b * Np + p];
}

// ---------------------------------------------------------------------------
// Host-side orchestration
// ---------------------------------------------------------------------------
static inline int cdiv(int a, int b) { return (a + b - 1) / b; }

static void launch_conv(hipStream_t s, const float* x, const _Float16* w, const void* b,
                        float* y, int B, int Cin, int H, int W, int Cout,
                        int k, int pad, int relu)
{
    int Ho = H + 2 * pad - k + 1, Wo = W + 2 * pad - k + 1;
    dim3 g(cdiv(Ho * Wo, BLOCK_PIX), cdiv(Cout, TILE_MBLK), B);
    if (k == 5)
        conv2d_wmma<5, 0><<<g, CONV_BLOCK, 0, s>>>(x, w, (const float*)b, y,
                                                   Cin, H, W, Cout, Ho, Wo, relu);
    else if (k == 3)
        conv2d_wmma<3, 1><<<g, CONV_BLOCK, 0, s>>>(x, w, (const float*)b, y,
                                                   Cin, H, W, Cout, Ho, Wo, relu);
    else
        conv2d_wmma<1, 0><<<g, CONV_BLOCK, 0, s>>>(x, w, (const float*)b, y,
                                                   Cin, H, W, Cout, Ho, Wo, relu);
}

// depth-9 chain: 8x (conv+relu, width 100) + final linear conv to 441 channels
static void run_chain(hipStream_t s, void* const* din, _Float16* const* w16, int base,
                      const float* x, int B, int cin, int H, int W, int k, int pad,
                      float* bufA, float* bufB, float* finalDst, int* Ho, int* Wo)
{
    const float* cur = x;
    float* dst = bufA;
    int c = cin;
    for (int i = 0; i < 9; ++i) {
        int co   = (i == 8) ? 441 : 100;
        float* d = (i == 8) ? finalDst : dst;
        launch_conv(s, cur, w16[base + 2 * i], din[base + 2 * i + 1], d,
                    B, c, H, W, co, k, pad, (i < 8) ? 1 : 0);
        H = H + 2 * pad - k + 1; W = W + 2 * pad - k + 1;
        cur = d;
        dst = (d == bufA) ? bufB : bufA;
        c = co;
    }
    *Ho = H; *Wo = W;
}

// simple_unet -> uout[B,51,H,W]
static void run_unet(hipStream_t s, void* const* din, _Float16* const* w16, int base,
                     const float* x, int B, int Cin, int H, int W,
                     float* h1, float* t0, float* t1, float* cat, float* uout)
{
    launch_conv(s, x,  w16[base + 0],  din[base + 1],  t0, B, Cin, H, W, 64, 3, 1, 1);   // c1
    launch_conv(s, t0, w16[base + 2],  din[base + 3],  h1, B, 64,  H, W, 64, 3, 1, 1);   // c2
    int Hh = H >> 1, Wh = W >> 1;
    { int n = B * 64 * Hh * Wh;
      maxpool2<<<cdiv(n, 256), 256, 0, s>>>(h1, t0, B, 64, H, W); }
    launch_conv(s, t0, w16[base + 4],  din[base + 5],  t1, B, 64,  Hh, Wh, 128, 3, 1, 1); // c3
    launch_conv(s, t1, w16[base + 6],  din[base + 7],  t0, B, 128, Hh, Wh, 128, 3, 1, 1); // c4
    { int n = B * 128 * H * W;
      upsample2<<<cdiv(n, 256), 256, 0, s>>>(t0, t1, B, 128, H, W); }
    launch_conv(s, t1, w16[base + 12], din[base + 13], t0, B, 128, H, W, 64, 3, 1, 1);    // up
    { int n = B * 64 * H * W;
      copy_crop<<<cdiv(n, 256), 256, 0, s>>>(t0, cat, B, 64, H, W, 0, 64, 0, 0, 128, H, W, 0);
      copy_crop<<<cdiv(n, 256), 256, 0, s>>>(h1, cat, B, 64, H, W, 0, 64, 0, 0, 128, H, W, 64); }
    launch_conv(s, cat, w16[base + 8],  din[base + 9],  t1, B, 128, H, W, 64, 3, 1, 1);   // c5
    launch_conv(s, t1,  w16[base + 10], din[base + 11], uout, B, 64, H, W, 51, 1, 0, 0);  // out
}

extern "C" void kernel_launch(void* const* d_in, const int* in_sizes, int n_in,
                              void* d_out, int out_size, void* d_ws, size_t ws_size,
                              hipStream_t stream)
{
    (void)in_sizes; (void)n_in; (void)out_size; (void)ws_size;

    // Param flatten order (jax pytree, dict keys sorted):
    //   g_diffuse[9x(w,b)] g_specular p_diffuse p_specular
    //   unet_d{c1,c2,c3,c4,c5,out,up}(w,b)  unet_s{...}
    const int GD = 7, GS = 25, PD = 43, PS = 61, UD = 79, US = 93;

    const int B = 2, H0 = 160, W0 = 160;
    const int Np0 = H0 * W0;
    float* out = (float*)d_out;

    size_t off = 0;
    auto alloc = [&](size_t elems) -> float* {
        float* p = (float*)((char*)d_ws + off);
        off = (off + elems * sizeof(float) + 255) & ~(size_t)255;
        return p;
    };

    // ---- pre-pack all conv weights to f16 (re-done each call; deterministic) ----
    _Float16* w16[112] = {};
    auto packW = [&](int idx, int cout, int cin, int k) {
        size_t nels = (size_t)cout * cin * k * k;
        _Float16* p = (_Float16*)((char*)d_ws + off);
        off = (off + nels * sizeof(_Float16) + 255) & ~(size_t)255;
        pack_f16<<<cdiv((int)nels, 256), 256, 0, stream>>>((const float*)d_in[idx], p, (int)nels);
        w16[idx] = p;
    };
    for (int c2 = 0; c2 < 2; ++c2) {
        int base = c2 ? GS : GD;
        for (int i = 0; i < 9; ++i) packW(base + 2 * i, (i == 8) ? 441 : 100, (i == 0) ? 27 : 100, 5);
    }
    for (int c2 = 0; c2 < 2; ++c2) {
        int base = c2 ? PS : PD;
        for (int i = 0; i < 9; ++i) packW(base + 2 * i, (i == 8) ? 441 : 100, (i == 0) ? 50 : 100, 3);
    }
    for (int u = 0; u < 2; ++u) {
        int base = u ? US : UD;
        packW(base + 0, 64, 27, 3);   packW(base + 2, 64, 64, 3);
        packW(base + 4, 128, 64, 3);  packW(base + 6, 128, 128, 3);
        packW(base + 8, 64, 128, 3);  packW(base + 10, 51, 64, 1);
        packW(base + 12, 64, 128, 3);
    }

    const int Hg = 124, Wg = 124, Np1 = Hg * Wg, crop = (H0 - Hg) / 2; // 18
    float* gin_d  = alloc((size_t)B * 27 * Np0);
    float* gin_s  = alloc((size_t)B * 27 * Np0);
    float* bufA   = alloc((size_t)B * 441 * Np1);   // ping (also softmax tmp)
    float* bufB   = alloc((size_t)B * 441 * Np1);   // pong
    float* kg_d   = alloc((size_t)B * 441 * Np1);
    float* kg_s   = alloc((size_t)B * 441 * Np1);
    float* kd     = alloc((size_t)B * 441 * Np1);
    float* ks     = alloc((size_t)B * 441 * Np1);
    float* b_d    = alloc((size_t)B * 3 * Np1);
    float* b_s    = alloc((size_t)B * 3 * Np1);
    float* in_d   = alloc((size_t)B * 27 * Np1);
    float* in_s   = alloc((size_t)B * 27 * Np1);
    float* tgt_d  = alloc((size_t)B * 27 * Np0);
    float* tgt_s  = alloc((size_t)B * 27 * Np0);
    float* feat_d = alloc((size_t)B * 50 * Np1);
    float* feat_s = alloc((size_t)B * 50 * Np1);
    float* h1     = alloc((size_t)B * 64 * Np0);
    float* t0     = alloc((size_t)B * 128 * Np0);
    float* t1     = alloc((size_t)B * 128 * Np0);
    float* cat    = alloc((size_t)B * 128 * Np0);
    float* uout   = alloc((size_t)B * 51 * Np0);

    float* o_final = out;
    float* o_rd    = o_final + (size_t)B * 3 * Np1;
    float* o_rs    = o_rd    + (size_t)B * 3 * Np1;
    float* o_gfin  = o_rs    + (size_t)B * 3 * Np1;
    float* o_grd   = o_gfin  + (size_t)B * 3 * Np1;
    float* o_grs   = o_grd   + (size_t)B * 3 * Np1;
    float* o_faked = o_grs   + (size_t)B * 3 * Np1;
    float* o_fakes = o_faked + (size_t)B * Np1;
    float* o_reald = o_fakes + (size_t)B * Np1;
    float* o_reals = o_reald + (size_t)B * Np0;

    int n, Ht, Wt;

    // slice first 27 channels of the 34-channel inputs
    n = B * 27 * Np0;
    copy_crop<<<cdiv(n, 256), 256, 0, stream>>>((const float*)d_in[0], gin_d, B, 34, H0, W0, 0, 27, 0, 0, 27, H0, W0, 0);
    copy_crop<<<cdiv(n, 256), 256, 0, stream>>>((const float*)d_in[1], gin_s, B, 34, H0, W0, 0, 27, 0, 0, 27, H0, W0, 0);

    // g-buffer kernel chains (5x5 VALID, 160 -> 124)
    run_chain(stream, d_in, w16, GD, gin_d, B, 27, H0, W0, 5, 0, bufA, bufB, kg_d, &Ht, &Wt);
    run_chain(stream, d_in, w16, GS, gin_s, B, 27, H0, W0, 5, 0, bufA, bufB, kg_s, &Ht, &Wt);

    // cropped radiance buffers
    n = B * 3 * Np1;
    copy_crop<<<cdiv(n, 256), 256, 0, stream>>>((const float*)d_in[2], b_d, B, 3, H0, W0, 0, 3, crop, crop, 3, Hg, Wg, 0);
    copy_crop<<<cdiv(n, 256), 256, 0, stream>>>((const float*)d_in[3], b_s, B, 3, H0, W0, 0, 3, crop, crop, 3, Hg, Wg, 0);

    // g-buffer kernel apply + combine
    n = B * Np1;
    kernel_apply_k<<<cdiv(n, 256), 256, 0, stream>>>(b_d, kg_d, o_grd, B, Hg, Wg);
    kernel_apply_k<<<cdiv(n, 256), 256, 0, stream>>>(b_s, kg_s, o_grs, B, Hg, Wg);
    n = B * 3 * Np1;
    combine_radiance<<<cdiv(n, 256), 256, 0, stream>>>((const float*)d_in[4], H0, W0, crop, o_grd, o_grs, o_gfin, B, Hg, Wg);

    // discriminator inputs
    n = B * 3 * Np1;
    copy_crop<<<cdiv(n, 256), 256, 0, stream>>>(o_grd, in_d, B, 3, Hg, Wg, 0, 3, 0, 0, 27, Hg, Wg, 0);
    copy_crop<<<cdiv(n, 256), 256, 0, stream>>>(o_grs, in_s, B, 3, Hg, Wg, 0, 3, 0, 0, 27, Hg, Wg, 0);
    n = B * 24 * Np1;
    copy_crop<<<cdiv(n, 256), 256, 0, stream>>>((const float*)d_in[0], in_d, B, 34, H0, W0, 10, 24, crop, crop, 27, Hg, Wg, 3);
    copy_crop<<<cdiv(n, 256), 256, 0, stream>>>((const float*)d_in[1], in_s, B, 34, H0, W0, 10, 24, crop, crop, 27, Hg, Wg, 3);
    n = B * 3 * Np0;
    copy_crop<<<cdiv(n, 256), 256, 0, stream>>>((const float*)d_in[5], tgt_d, B, 3, H0, W0, 0, 3, 0, 0, 27, H0, W0, 0);
    copy_crop<<<cdiv(n, 256), 256, 0, stream>>>((const float*)d_in[6], tgt_s, B, 3, H0, W0, 0, 3, 0, 0, 27, H0, W0, 0);
    n = B * 24 * Np0;
    copy_crop<<<cdiv(n, 256), 256, 0, stream>>>((const float*)d_in[0], tgt_d, B, 34, H0, W0, 10, 24, 0, 0, 27, H0, W0, 3);
    copy_crop<<<cdiv(n, 256), 256, 0, stream>>>((const float*)d_in[1], tgt_s, B, 34, H0, W0, 10, 24, 0, 0, 27, H0, W0, 3);

    // unet_d: fake (124^2) then real (160^2)
    run_unet(stream, d_in, w16, UD, in_d, B, 27, Hg, Wg, h1, t0, t1, cat, uout);
    n = B * Np1;
    sigmoid_ch0<<<cdiv(n, 256), 256, 0, stream>>>(uout, o_faked, B, 51, Np1);
    n = B * 50 * Np1;
    copy_crop<<<cdiv(n, 256), 256, 0, stream>>>(uout, feat_d, B, 51, Hg, Wg, 1, 50, 0, 0, 50, Hg, Wg, 0);
    run_unet(stream, d_in, w16, UD, tgt_d, B, 27, H0, W0, h1, t0, t1, cat, uout);
    n = B * Np0;
    sigmoid_ch0<<<cdiv(n, 256), 256, 0, stream>>>(uout, o_reald, B, 51, Np0);

    // unet_s
    run_unet(stream, d_in, w16, US, in_s, B, 27, Hg, Wg, h1, t0, t1, cat, uout);
    n = B * Np1;
    sigmoid_ch0<<<cdiv(n, 256), 256, 0, stream>>>(uout, o_fakes, B, 51, Np1);
    n = B * 50 * Np1;
    copy_crop<<<cdiv(n, 256), 256, 0, stream>>>(uout, feat_s, B, 51, Hg, Wg, 1, 50, 0, 0, 50, Hg, Wg, 0);
    run_unet(stream, d_in, w16, US, tgt_s, B, 27, H0, W0, h1, t0, t1, cat, uout);
    n = B * Np0;
    sigmoid_ch0<<<cdiv(n, 256), 256, 0, stream>>>(uout, o_reals, B, 51, Np0);

    // feature kernel chains (3x3 SAME at 124^2)
    run_chain(stream, d_in, w16, PD, feat_d, B, 50, Hg, Wg, 3, 1, bufA, bufB, kd, &Ht, &Wt);
    run_chain(stream, d_in, w16, PS, feat_s, B, 50, Hg, Wg, 3, 1, bufA, bufB, ks, &Ht, &Wt);

    // k = k + k_gbuf * fake
    n = B * 441 * Np1;
    kernel_fma_fake<<<cdiv(n, 256), 256, 0, stream>>>(kd, kg_d, o_faked, B, 441, Np1);
    kernel_fma_fake<<<cdiv(n, 256), 256, 0, stream>>>(ks, kg_s, o_fakes, B, 441, Np1);

    // faithful double-softmax: softmax(k) then KernelApply(softmax=True)
    n = B * Np1;
    softmax_ch<<<cdiv(n, 256), 256, 0, stream>>>(kd, bufA, B, 441, Np1);
    kernel_apply_k<<<cdiv(n, 256), 256, 0, stream>>>(b_d, bufA, o_rd, B, Hg, Wg);
    softmax_ch<<<cdiv(n, 256), 256, 0, stream>>>(ks, bufA, B, 441, Np1);
    kernel_apply_k<<<cdiv(n, 256), 256, 0, stream>>>(b_s, bufA, o_rs, B, Hg, Wg);

    // final radiance
    n = B * 3 * Np1;
    combine_radiance<<<cdiv(n, 256), 256, 0, stream>>>((const float*)d_in[4], H0, W0, crop, o_rd, o_rs, o_final, B, Hg, Wg);
}
